// CrossAttention_36309653520611
// MI455X (gfx1250) — compile-verified, matrix operand order checked by reading, not run on hardware
//
#include <hip/hip_runtime.h>
#include <hip/hip_bf16.h>

// ---------------------------------------------------------------------------
// Types for CDNA5 WMMA (wave32): v_wmma_f32_16x16x32_bf16
// ---------------------------------------------------------------------------
typedef __bf16 bf16;
typedef bf16  v16bf __attribute__((ext_vector_type(16)));
typedef bf16  v8bf  __attribute__((ext_vector_type(8)));
typedef float v8f   __attribute__((ext_vector_type(8)));
typedef unsigned int u32;
typedef u32 u32x4 __attribute__((ext_vector_type(4)));
typedef int  i32x4 __attribute__((ext_vector_type(4)));
typedef int  i32x8 __attribute__((ext_vector_type(8)));

#define B_   16
#define S_   512
#define DIM_ 1024
#define H_   16
#define HD_  64
#define NTOK (B_ * S_ * DIM_)   // 8,388,608 elements in [B,S,DIM]
#define NW   (DIM_ * DIM_)      // 1,048,576 per weight

// 16-bit A/B fragment load (ISA 7.12.2): row = 32 contiguous K elems.
// lane<16 -> K chunks {0..7, 16..23}; lane>=16 -> {8..15, 24..31}
__device__ __forceinline__ v16bf load_frag_row(const bf16* row, int half) {
  const v8bf* p = reinterpret_cast<const v8bf*>(row);
  union { v16bf v; v8bf h[2]; } u;
  u.h[0] = p[half];
  u.h[1] = p[half + 2];
  return u.v;
}

__device__ __forceinline__ v8f wmma_bf16(v16bf a, v16bf b, v8f c) {
  return __builtin_amdgcn_wmma_f32_16x16x32_bf16(
      /*neg_a=*/false, a, /*neg_b=*/false, b,
      /*c_mod=*/(short)0, c, /*reuse_a=*/false, /*reuse_b=*/false);
}

// ---------------------------------------------------------------------------
// Tensor Data Mover: 2-D bf16 tile (tile_rows x 32 elems) from a row-major
// [tensor_rows x 1024] bf16 matrix into contiguous LDS. Descriptor packing
// per CDNA5 ISA (group0) / §8.4 (group1).
// Issued by a single wave; tracked with TENSORcnt.
// ---------------------------------------------------------------------------
__device__ __forceinline__ void tdm_load_tile_2d(const bf16* gsrc, u32 lds_off,
                                                 u32 tensor_rows, u32 tile_rows) {
  unsigned long long ga = (unsigned long long)(const void*)gsrc;
  u32x4 g0;
  g0[0] = 1u;                                    // count=1, user mode
  g0[1] = lds_off;                               // lds_addr [63:32]
  g0[2] = (u32)(ga & 0xffffffffu);               // global_addr lo
  g0[3] = (u32)((ga >> 32) & 0x01ffffffu)        // global_addr hi (bits 56:32)
          | (2u << 30);                          // type=2 ("image")
  i32x8 g1;
  g1[0] = (int)(1u << 16);                       // data_size=1 -> 2 bytes/elem
  g1[1] = (int)((DIM_ & 0xffffu) << 16);         // tensor_dim0 lo16
  g1[2] = (int)((DIM_ >> 16) | ((tensor_rows & 0xffffu) << 16)); // dim0 hi / dim1 lo
  g1[3] = (int)((tensor_rows >> 16) | (32u << 16));              // dim1 hi / tile_dim0=32
  g1[4] = (int)(tile_rows & 0xffffu);            // tile_dim1 (tile_dim2=0)
  g1[5] = (int)DIM_;                             // tensor_dim0_stride lo32
  g1[6] = 0;                                     // stride0 hi16 / stride1 lo16
  g1[7] = 0;                                     // stride1 hi32
  i32x4 z4 = {0, 0, 0, 0};
#if defined(__clang_major__) && (__clang_major__ >= 23)
  i32x8 z8 = {0, 0, 0, 0, 0, 0, 0, 0};
  __builtin_amdgcn_tensor_load_to_lds(g0, g1, z4, z4, z8, 0);
#else
  __builtin_amdgcn_tensor_load_to_lds(g0, g1, z4, z4, 0);
#endif
}

// ---------------------------------------------------------------------------
// fp32 -> bf16 elementwise convert
// ---------------------------------------------------------------------------
__global__ void cvt_bf16_kernel(const float* __restrict__ src,
                                bf16* __restrict__ dst, int n) {
  int i = blockIdx.x * blockDim.x + threadIdx.x;
  if (i < n) dst[i] = (bf16)src[i];
}

// fp32 [1024][1024] -> bf16 transposed [1024][1024] (WT[n][k] = W[k][n])
__global__ void transpose_cvt_kernel(const float* __restrict__ W,
                                     bf16* __restrict__ WT) {
  __shared__ float tile[32][33];
  int tx = threadIdx.x & 31, ty = threadIdx.x >> 5;  // 32 x 8
  int c0 = blockIdx.x * 32, r0 = blockIdx.y * 32;
#pragma unroll
  for (int r = 0; r < 32; r += 8)
    tile[ty + r][tx] = W[(size_t)(r0 + ty + r) * DIM_ + c0 + tx];
  __syncthreads();
#pragma unroll
  for (int r = 0; r < 32; r += 8)
    WT[(size_t)(c0 + ty + r) * DIM_ + r0 + tx] = (bf16)tile[tx][ty + r];
}

// ---------------------------------------------------------------------------
// GEMM: C[M,1024] = A[M,1024](bf16) x W[1024,1024], W given transposed (WT).
// 256 threads = 8 waves; block tile 128x128, BK=32. Each wave: 32x64 (2x4
// WMMA tiles). A/B tiles are DMA'd into double-buffered LDS by the Tensor
// Data Mover (wave 0 issues, TENSORcnt pipelined one tile ahead).
// ---------------------------------------------------------------------------
__global__ void __launch_bounds__(256)
gemm_bf16_kernel(const bf16* __restrict__ A, const bf16* __restrict__ WT,
                 float* __restrict__ C, const float* __restrict__ bias) {
  __shared__ __align__(16) bf16 As[2][128 * 32];   // 2 x 8 KB
  __shared__ __align__(16) bf16 Bs[2][128 * 32];   // 2 x 8 KB

  const int tid  = threadIdx.x;
  const int lane = tid & 31;
  const int wave = tid >> 5;
  const int wm   = wave & 3;   // M quadrant (rows wm*32)
  const int wn   = wave >> 2;  // N half    (cols wn*64)
  const int mBase = blockIdx.y * 128;
  const int nBase = blockIdx.x * 128;
  const int ln15 = lane & 15;
  const int half = lane >> 4;
  const int NITER = DIM_ / 32;

  v8f acc[2][4] = {};

  // Prologue: TDM-load tile 0 into buffer 0 (wave 0 only; EXEC-skipped waves
  // branch around; TDM itself ignores EXEC and issues once per wave).
  if (wave == 0) {
    tdm_load_tile_2d(A  + (size_t)mBase * DIM_, (u32)(size_t)&As[0][0], B_ * S_, 128);
    tdm_load_tile_2d(WT + (size_t)nBase * DIM_, (u32)(size_t)&Bs[0][0], DIM_, 128);
  }

  for (int i = 0; i < NITER; ++i) {
    const int buf = i & 1;
    if (wave == 0) {
      if (i + 1 < NITER) {
        // Pipeline: fetch tile i+1 into the other buffer, then wait for the
        // two oldest TDM ops (tile i) -- TDM completes in order per wave.
        const int k1 = (i + 1) * 32;
        tdm_load_tile_2d(A  + (size_t)mBase * DIM_ + k1, (u32)(size_t)&As[buf ^ 1][0],
                         B_ * S_, 128);
        tdm_load_tile_2d(WT + (size_t)nBase * DIM_ + k1, (u32)(size_t)&Bs[buf ^ 1][0],
                         DIM_, 128);
        __builtin_amdgcn_s_wait_tensorcnt(2);
      } else {
        __builtin_amdgcn_s_wait_tensorcnt(0);
      }
    }
    __syncthreads();  // tile i visible to all waves

    v16bf af[2];
#pragma unroll
    for (int mi = 0; mi < 2; ++mi)
      af[mi] = load_frag_row(&As[buf][(wm * 32 + mi * 16 + ln15) * 32], half);
#pragma unroll
    for (int ni = 0; ni < 4; ++ni) {
      v16bf bf = load_frag_row(&Bs[buf][(wn * 64 + ni * 16 + ln15) * 32], half);
#pragma unroll
      for (int mi = 0; mi < 2; ++mi)
        acc[mi][ni] = wmma_bf16(af[mi], bf, acc[mi][ni]);
    }
    __syncthreads();  // tile i consumed before TDM overwrites this buffer
  }

  // Epilogue: C layout (7.12.2): VGPR r -> M = r (+8 for upper lanes), N = lane&15
  const int rbase = half * 8;
#pragma unroll
  for (int mi = 0; mi < 2; ++mi)
#pragma unroll
    for (int ni = 0; ni < 4; ++ni) {
      int col = nBase + wn * 64 + ni * 16 + ln15;
      float bv = bias ? bias[col] : 0.0f;
#pragma unroll
      for (int r = 0; r < 8; ++r) {
        int row = mBase + wm * 32 + mi * 16 + rbase + r;
        C[(size_t)row * DIM_ + col] = acc[mi][ni][r] + bv;
      }
    }
}

// ---------------------------------------------------------------------------
// RoPE (tiled cos/sin tables, adjacent-pair rotation — exactly as reference)
// + repack f32 [B,S,H*HD] -> bf16 [B,H,S,HD]. One thread per (b,s,h,pair).
// ---------------------------------------------------------------------------
__global__ void rope_pack_kernel(const float* __restrict__ P,
                                 bf16* __restrict__ out, int applyRope) {
  int idx = blockIdx.x * blockDim.x + threadIdx.x;  // B*S*H*32 = 4,194,304
  int i = idx & 31;
  int h = (idx >> 5) & 15;
  int s = (idx >> 9) & 511;
  int b = idx >> 18;
  const float* src = P + ((size_t)b * S_ + s) * DIM_ + h * HD_;
  float x0 = src[2 * i], x1 = src[2 * i + 1];
  float o0 = x0, o1 = x1;
  if (applyRope) {
    float t = (float)s;
    float f0 = __powf(10000.0f, -(float)((2 * i) & 31) / 32.0f);
    float f1 = __powf(10000.0f, -(float)((2 * i + 1) & 31) / 32.0f);
    float c0 = __cosf(t * f0), s0 = __sinf(t * f0);
    float c1 = __cosf(t * f1), s1 = __sinf(t * f1);
    o0 = x0 * c0 - x1 * s0;   // rot[2i]   = -x[2i+1]
    o1 = x1 * c1 + x0 * s1;   // rot[2i+1] =  x[2i]
  }
  bf16* dst = out + (((size_t)b * H_ + h) * S_ + s) * HD_ + 2 * i;
  dst[0] = (bf16)o0;
  dst[1] = (bf16)o1;
}

// ---------------------------------------------------------------------------
// Flash attention. Block = (b*16+h, q-tile of 128 rows). 8 waves; wave w owns
// score rows [w*16, w*16+16) over all 64 KV columns of a block -> row-wise
// softmax stays wave-local (16-lane shfl reductions). Output written back as
// bf16 [B,Sq,H*HD] so the output projection reuses the GEMM kernel.
// ---------------------------------------------------------------------------
__global__ void __launch_bounds__(256)
flash_attn_kernel(const bf16* __restrict__ q, const bf16* __restrict__ k,
                  const bf16* __restrict__ v, bf16* __restrict__ attn) {
  __shared__ __align__(16) bf16 Qs[128 * 64];   // [qrow][d]
  __shared__ __align__(16) bf16 Ks[64 * 64];    // [kvrow][d]
  __shared__ __align__(16) bf16 Vst[64 * 64];   // transposed: [d][kvrow]
  __shared__ __align__(16) bf16 Ps[8 * 16 * 64];// wave-private P tiles

  const int tid  = threadIdx.x;
  const int lane = tid & 31;
  const int wave = tid >> 5;
  const int ln15 = lane & 15;
  const int half = lane >> 4;
  const int bh = blockIdx.x;                 // b*16 + h
  const int q0 = blockIdx.y * 128;
  const size_t hoff = (size_t)bh * S_ * HD_;

  // stage Q tile
#pragma unroll
  for (int i = 0; i < 4; ++i) {
    int c = tid + i * 256;                   // 1024 chunks of 8
    int m = c >> 3, dc = (c & 7) * 8;
    *(v8bf*)(Qs + m * 64 + dc) =
        *(const v8bf*)(q + hoff + (size_t)(q0 + m) * HD_ + dc);
  }
  __syncthreads();

  const bf16* qrow = Qs + (wave * 16 + ln15) * 64;
  v16bf qf0 = load_frag_row(qrow, half);        // d 0..31
  v16bf qf1 = load_frag_row(qrow + 32, half);   // d 32..63

  float m_r[8], l_r[8];
#pragma unroll
  for (int r = 0; r < 8; ++r) { m_r[r] = -1e30f; l_r[r] = 0.0f; }
  v8f oacc[4] = {};
  const float scale = 0.125f;                   // HD^-0.5

  for (int j = 0; j < S_; j += 64) {
    __syncthreads();
    // stage K block (row-major) and V block (transposed)
#pragma unroll
    for (int i = 0; i < 2; ++i) {
      int c = tid + i * 256;
      int kr = c >> 3, dc = (c & 7) * 8;
      *(v8bf*)(Ks + kr * 64 + dc) =
          *(const v8bf*)(k + hoff + (size_t)(j + kr) * HD_ + dc);
    }
#pragma unroll
    for (int i = 0; i < 16; ++i) {
      int idx = tid + i * 256;
      int kr = idx >> 6, d = idx & 63;
      Vst[d * 64 + kr] = v[hoff + (size_t)(j + kr) * HD_ + d];
    }
    __syncthreads();

    // scores S = Q x K^T  (B-frag column = KV row; K-dim = head dim)
    v8f s[4];
#pragma unroll
    for (int ni = 0; ni < 4; ++ni) {
      const bf16* krow = Ks + (ni * 16 + ln15) * 64;
      v16bf b0 = load_frag_row(krow, half);
      v16bf b1 = load_frag_row(krow + 32, half);
      v8f z = {};
      z = wmma_bf16(qf0, b0, z);
      z = wmma_bf16(qf1, b1, z);
      s[ni] = z;
    }
#pragma unroll
    for (int ni = 0; ni < 4; ++ni)
#pragma unroll
      for (int r = 0; r < 8; ++r) s[ni][r] *= scale;

    // online softmax (lane tracks row rbase+r of its wave tile)
#pragma unroll
    for (int r = 0; r < 8; ++r) {
      float mx = -1e30f;
#pragma unroll
      for (int ni = 0; ni < 4; ++ni) mx = fmaxf(mx, s[ni][r]);
      for (int d = 1; d < 16; d <<= 1) mx = fmaxf(mx, __shfl_xor(mx, d, 32));
      float mnew = fmaxf(m_r[r], mx);
      float corr = __expf(m_r[r] - mnew);
      float rsum = 0.0f;
#pragma unroll
      for (int ni = 0; ni < 4; ++ni) {
        float p = __expf(s[ni][r] - mnew);
        s[ni][r] = p;
        rsum += p;
      }
      for (int d = 1; d < 16; d <<= 1) rsum += __shfl_xor(rsum, d, 32);
      l_r[r] = l_r[r] * corr + rsum;
      m_r[r] = mnew;
#pragma unroll
      for (int ni = 0; ni < 4; ++ni) oacc[ni][r] *= corr;
    }

    // C-layout -> A-frag layout via wave-private LDS round trip
    bf16* pw = Ps + wave * 16 * 64;
    const int rbase = half * 8;
#pragma unroll
    for (int ni = 0; ni < 4; ++ni)
#pragma unroll
      for (int r = 0; r < 8; ++r)
        pw[(rbase + r) * 64 + ni * 16 + ln15] = (bf16)s[ni][r];

    const bf16* prow = pw + ln15 * 64;
    v16bf p0 = load_frag_row(prow, half);        // kv 0..31
    v16bf p1 = load_frag_row(prow + 32, half);   // kv 32..63

    // O += P x V (B-frag column = d; K-dim = kv, from transposed Vst)
#pragma unroll
    for (int ni = 0; ni < 4; ++ni) {
      const bf16* vrow = Vst + (ni * 16 + ln15) * 64;
      v16bf vb0 = load_frag_row(vrow, half);
      v16bf vb1 = load_frag_row(vrow + 32, half);
      oacc[ni] = wmma_bf16(p0, vb0, oacc[ni]);
      oacc[ni] = wmma_bf16(p1, vb1, oacc[ni]);
    }
  }

  // normalize and store bf16 [B, Sq, H*HD]
  const int b = bh >> 4, h = bh & 15;
  const int rbase = half * 8;
#pragma unroll
  for (int r = 0; r < 8; ++r) {
    float inv = 1.0f / l_r[r];
    int row = q0 + wave * 16 + rbase + r;
#pragma unroll
    for (int ni = 0; ni < 4; ++ni) {
      int d = ni * 16 + ln15;
      attn[((size_t)b * S_ + row) * DIM_ + h * HD_ + d] =
          (bf16)(oacc[ni][r] * inv);
    }
  }
}

// ---------------------------------------------------------------------------
// Launcher: cvt -> (gemm + rope_pack) x3 -> flash -> gemm(+bias) -> d_out
// ---------------------------------------------------------------------------
extern "C" void kernel_launch(void* const* d_in, const int* in_sizes, int n_in,
                              void* d_out, int out_size, void* d_ws, size_t ws_size,
                              hipStream_t stream) {
  (void)in_sizes; (void)n_in; (void)out_size; (void)ws_size;
  const float* x   = (const float*)d_in[0];
  const float* ctx = (const float*)d_in[1];
  const float* Wq  = (const float*)d_in[2];
  const float* Wk  = (const float*)d_in[3];
  const float* Wv  = (const float*)d_in[4];
  const float* Wo  = (const float*)d_in[5];
  const float* bo  = (const float*)d_in[6];
  float* out = (float*)d_out;

  char* ws = (char*)d_ws;
  float* Pf32 = (float*)ws;                         // 32 MB
  bf16* xb    = (bf16*)(ws + (size_t)NTOK * 4);     // 16 MB
  bf16* cb    = xb + NTOK;                          // 16 MB
  bf16* WqT   = cb + NTOK;                          // 2 MB x4
  bf16* WkT   = WqT + NW;
  bf16* WvT   = WkT + NW;
  bf16* WoT   = WvT + NW;
  bf16* qb    = WoT + NW;                           // 16 MB x3
  bf16* kb    = qb + NTOK;
  bf16* vb    = kb + NTOK;
  bf16* attnb = vb + NTOK;                          // 16 MB

  dim3 gGemm(DIM_ / 128, (B_ * S_) / 128);          // (8, 64)
  dim3 gFlash(B_ * H_, S_ / 128);                   // (256, 4)
  const int ropeBlocks = (B_ * S_ * H_ * 32) / 256; // 16384

  cvt_bf16_kernel<<<NTOK / 256, 256, 0, stream>>>(x, xb, NTOK);
  cvt_bf16_kernel<<<NTOK / 256, 256, 0, stream>>>(ctx, cb, NTOK);
  transpose_cvt_kernel<<<dim3(32, 32), 256, 0, stream>>>(Wq, WqT);
  transpose_cvt_kernel<<<dim3(32, 32), 256, 0, stream>>>(Wk, WkT);
  transpose_cvt_kernel<<<dim3(32, 32), 256, 0, stream>>>(Wv, WvT);
  transpose_cvt_kernel<<<dim3(32, 32), 256, 0, stream>>>(Wo, WoT);

  gemm_bf16_kernel<<<gGemm, 256, 0, stream>>>(xb, WqT, Pf32, nullptr);
  rope_pack_kernel<<<ropeBlocks, 256, 0, stream>>>(Pf32, qb, 1);

  gemm_bf16_kernel<<<gGemm, 256, 0, stream>>>(cb, WkT, Pf32, nullptr);
  rope_pack_kernel<<<ropeBlocks, 256, 0, stream>>>(Pf32, kb, 1);

  gemm_bf16_kernel<<<gGemm, 256, 0, stream>>>(cb, WvT, Pf32, nullptr);
  rope_pack_kernel<<<ropeBlocks, 256, 0, stream>>>(Pf32, vb, 0);

  flash_attn_kernel<<<gFlash, 256, 0, stream>>>(qb, kb, vb, attnb);

  gemm_bf16_kernel<<<gGemm, 256, 0, stream>>>(attnb, WoT, out, bo);
}
